// BaseAttentionConvolution_53841710023032
// MI455X (gfx1250) — matrix operand order repaired and checked by reading, not compiled, not run on hardware
//
#include <hip/hip_runtime.h>

// ---------------------------------------------------------------------------
// Neighborhood attention (7x7 window) on [B=2, H=96, W=96, C=128], KD=64,
// OD=128, for MI455X (gfx1250, wave32, WMMA f16 16x16x32).
//  - Stage 1/3 GEMMs on v_wmma_f32_16x16x32_f16 (f32 accumulate)
//  - Stage 2 halos staged with global_load_async_to_lds_b128 (ASYNCcnt)
//  - All global accesses forced to address_space(1) -> global_load/store
// ---------------------------------------------------------------------------

typedef __attribute__((ext_vector_type(16))) _Float16 v16h;
typedef __attribute__((ext_vector_type(8)))  float    v8f;
typedef __attribute__((ext_vector_type(8)))  _Float16 h8;
typedef __attribute__((ext_vector_type(4)))  float    f32x4;
typedef __attribute__((ext_vector_type(4)))  unsigned u32x4;

#define GDEV __attribute__((address_space(1)))

#define BATCH 2
#define HH    96
#define WW    96
#define CC    128
#define KD    64
#define OD    128
#define MROWS (BATCH * HH * WW)   // 18432

// generic(shared) pointer -> LDS byte offset for ds/async instructions
__device__ __forceinline__ unsigned lds_offset(void* p) {
  return (unsigned)(uintptr_t)(__attribute__((address_space(3))) void*)p;
}

// async global -> LDS, 16B per lane, tracked on ASYNCcnt
__device__ __forceinline__ void async_b128(unsigned lds_dst,
                                           const GDEV void* gsrc) {
  asm volatile("global_load_async_to_lds_b128 %0, %1, off"
               :: "v"(lds_dst), "v"(gsrc)
               : "memory");
}
__device__ __forceinline__ void wait_async0() {
  asm volatile("s_wait_asynccnt 0" ::: "memory");
}

// ----------------------- WMMA fragment helpers -----------------------------
// A-matrix (16x32 f16, MxK), contiguous K order (matches load_bT):
//   lanes 0-15  hold row M=lane,    K in {0..15}
//   lanes 16-31 hold row M=lane-16, K in {16..31}
__device__ __forceinline__ v16h load_a_f32(const GDEV float* src, int ld,
                                           int lane) {
  const int r  = lane & 15;
  const int hi = lane >> 4;
  const GDEV f32x4* p0 = (const GDEV f32x4*)(src + (size_t)r * ld + 16 * hi);
  const GDEV f32x4* p1 =
      (const GDEV f32x4*)(src + (size_t)r * ld + 8 + 16 * hi);
  f32x4 a0 = p0[0], a1 = p0[1], b0 = p1[0], b1 = p1[1];
  v16h a;
#pragma unroll
  for (int e = 0; e < 4; ++e) {
    a[e]      = (_Float16)a0[e];
    a[4 + e]  = (_Float16)a1[e];
    a[8 + e]  = (_Float16)b0[e];
    a[12 + e] = (_Float16)b1[e];
  }
  return a;
}

__device__ __forceinline__ v16h load_a_f16(const GDEV _Float16* src, int ld,
                                           int lane) {
  const int r  = lane & 15;
  const int hi = lane >> 4;
  const GDEV h8* p0 = (const GDEV h8*)(src + (size_t)r * ld + 16 * hi);
  const GDEV h8* p1 = (const GDEV h8*)(src + (size_t)r * ld + 8 + 16 * hi);
  h8 x = *p0, y = *p1;
  v16h a;
#pragma unroll
  for (int i = 0; i < 8; ++i) { a[i] = x[i]; a[8 + i] = y[i]; }
  return a;
}

// B-matrix (32x16 f16, KxN): lane n (0-15) holds col N=n, K=0..15;
// lanes 16-31 hold col N=lane-16, K=16..31. Weights stored transposed
// (wT[n][k], ldk=128) so each lane reads one contiguous 32B v16h.
__device__ __forceinline__ v16h load_bT(const GDEV _Float16* wT, int ldk,
                                        int kc, int nt, int lane) {
  const int n  = (nt * 16) + (lane & 15);
  const int hi = lane >> 4;
  const GDEV _Float16* p = wT + (size_t)n * ldk + kc * 32 + hi * 16;
  return *(const GDEV v16h*)p;
}

// ----------------------- Stage 0: conversions -------------------------------
__global__ void cvt_f32_to_f16(const float* s_, _Float16* d_, int n) {
  const GDEV float* s = (const GDEV float*)s_;
  GDEV _Float16* d = (GDEV _Float16*)d_;
  int i = blockIdx.x * blockDim.x + threadIdx.x;
  if (i < n) d[i] = (_Float16)s[i];
}

// s is [K][N] row-major f32 -> d is [N][K] f16 (transposed)
__global__ void cvtT_f32_to_f16(const float* s_, _Float16* d_, int K, int N) {
  const GDEV float* s = (const GDEV float*)s_;
  GDEV _Float16* d = (GDEV _Float16*)d_;
  int i = blockIdx.x * blockDim.x + threadIdx.x;
  if (i < K * N) {
    int k = i / N, n = i % N;
    d[(size_t)n * K + k] = (_Float16)s[i];
  }
}

// ----------------------- Stage 1: q/k projections ---------------------------
// One 16-row M-tile per wave: N=64 (4 tiles) x K=128 (4 chunks of 32)
// -> 16 v_wmma_f32_16x16x32_f16 per projection.
__device__ __forceinline__ void proj_tile(const GDEV float* src,
                                          const GDEV _Float16* wT,
                                          const GDEV float* bias,
                                          GDEV _Float16* dst, int m0,
                                          int lane) {
  v8f acc[4] = {};
#pragma unroll
  for (int kc = 0; kc < 4; ++kc) {
    v16h a = load_a_f32(src + (size_t)m0 * CC + kc * 32, CC, lane);
#pragma unroll
    for (int nt = 0; nt < 4; ++nt) {
      v16h b = load_bT(wT, CC, kc, nt, lane);
      acc[nt] = __builtin_amdgcn_wmma_f32_16x16x32_f16(
          false, a, false, b, (short)0, acc[nt], false, false);
    }
  }
  const int n = lane & 15, hi = lane >> 4;
#pragma unroll
  for (int nt = 0; nt < 4; ++nt) {
    float bv = bias[nt * 16 + n];
#pragma unroll
    for (int i = 0; i < 8; ++i)
      dst[(size_t)(m0 + i + 8 * hi) * KD + nt * 16 + n] =
          (_Float16)(acc[nt][i] + bv);
  }
}

__global__ __launch_bounds__(128) void proj_qk_kernel(
    const float* Q, const float* Ksrc, const _Float16* wqT,
    const _Float16* wkT, const float* bq, const float* bk, _Float16* q_h,
    _Float16* k_h) {
  const int lane = threadIdx.x & 31;
  const int wave = blockIdx.x * 4 + (threadIdx.x >> 5);
  const int m0   = wave * 16;
  proj_tile((const GDEV float*)Q, (const GDEV _Float16*)wqT,
            (const GDEV float*)bq, (GDEV _Float16*)q_h, m0, lane);
  proj_tile((const GDEV float*)Ksrc, (const GDEV _Float16*)wkT,
            (const GDEV float*)bk, (GDEV _Float16*)k_h, m0, lane);
}

// ----------------------- Stage 2: neighborhood attention --------------------
// 8x16 pixel tile per 128-thread block, one pixel per thread.
// Phase A: k halo (14x22x64 f16) async-staged to LDS -> 49 scores, softmax.
// Phase B: V halo in 4 channel-chunks of 32 (14x22x32 f16), reusing LDS.
#define TH 8
#define TW 16
#define HALO_H (TH + 6)
#define HALO_W (TW + 6)
#define KP 72   // padded halfs per halo pixel (144B, 16B aligned)
#define VP 40   // padded halfs per halo pixel for a 32-channel V chunk (80B)

__global__ __launch_bounds__(128) void natten_kernel(
    const _Float16* q_h_, const _Float16* k_h_, const _Float16* v_h_,
    _Float16* att_h_) {
  __shared__ __align__(16) _Float16 smem[HALO_H * HALO_W * KP];

  const GDEV _Float16* q_h = (const GDEV _Float16*)q_h_;
  const GDEV _Float16* k_h = (const GDEV _Float16*)k_h_;
  const GDEV _Float16* v_h = (const GDEV _Float16*)v_h_;
  GDEV _Float16* att_h = (GDEV _Float16*)att_h_;

  const int tid = threadIdx.x;
  const int tx = tid & 15, ty = tid >> 4;
  const int w0 = blockIdx.x * TW, h0 = blockIdx.y * TH, b = blockIdx.z;
  const int gh = h0 + ty, gw = w0 + tx;
  const size_t pix = (size_t)(b * HH + gh) * WW + gw;
  const float NEG_INF = -__builtin_inff();

  // ---- stage k halo into LDS via async copies (OOB -> ds zero-fill) ----
  for (int idx = tid; idx < HALO_H * HALO_W; idx += 128) {
    int rr = idx / HALO_W, cc = idx % HALO_W;
    int sh = h0 + rr - 3, sw = w0 + cc - 3;
    _Float16* dl = smem + idx * KP;
    if (sh >= 0 && sh < HH && sw >= 0 && sw < WW) {
      const GDEV _Float16* src = k_h + ((size_t)(b * HH + sh) * WW + sw) * KD;
      const unsigned doff = lds_offset(dl);
#pragma unroll
      for (int i = 0; i < 8; ++i) async_b128(doff + i * 16, src + i * 8);
    } else {
      u32x4 z = {0u, 0u, 0u, 0u};
      u32x4* d = (u32x4*)dl;
#pragma unroll
      for (int i = 0; i < 8; ++i) d[i] = z;
    }
  }

  // ---- q row into registers (overlaps with async halo fill) ----
  float qf[KD];
  {
    const GDEV h8* qp = (const GDEV h8*)(q_h + pix * KD);
#pragma unroll
    for (int i = 0; i < 8; ++i) {
      h8 v = qp[i];
#pragma unroll
      for (int e = 0; e < 8; ++e) qf[i * 8 + e] = (float)v[e];
    }
  }
  wait_async0();
  __syncthreads();

  // ---- 49 scores (masked) ----
  float s[49];
#pragma unroll
  for (int iy = 0; iy < 7; ++iy) {
#pragma unroll
    for (int ix = 0; ix < 7; ++ix) {
      const int sh = gh + iy - 3, sw = gw + ix - 3;
      if (sh < 0 || sh >= HH || sw < 0 || sw >= WW) {
        s[iy * 7 + ix] = NEG_INF;
        continue;
      }
      const h8* kp = (const h8*)(smem + ((ty + iy) * HALO_W + (tx + ix)) * KP);
      float acc = 0.f;
#pragma unroll
      for (int i = 0; i < 8; ++i) {
        h8 t = kp[i];
#pragma unroll
        for (int e = 0; e < 8; ++e) acc += qf[i * 8 + e] * (float)t[e];
      }
      s[iy * 7 + ix] = acc * 0.125f;  // / sqrt(KD)
    }
  }

  // ---- softmax over the 49-window ----
  float m = NEG_INF;
#pragma unroll
  for (int t = 0; t < 49; ++t) m = fmaxf(m, s[t]);
  float sum = 0.f;
#pragma unroll
  for (int t = 0; t < 49; ++t) {
    float p = exp2f((s[t] - m) * 1.44269504088896f);  // exp(-inf) -> 0
    s[t] = p;
    sum += p;
  }
  const float inv = 1.f / sum;

  // ---- V aggregation, 32 channels per chunk, reusing LDS ----
  for (int ch = 0; ch < 4; ++ch) {
    __syncthreads();  // everyone done reading previous LDS contents
    for (int idx = tid; idx < HALO_H * HALO_W; idx += 128) {
      int rr = idx / HALO_W, cc = idx % HALO_W;
      int sh = h0 + rr - 3, sw = w0 + cc - 3;
      _Float16* dl = smem + idx * VP;
      if (sh >= 0 && sh < HH && sw >= 0 && sw < WW) {
        const GDEV _Float16* src =
            v_h + ((size_t)(b * HH + sh) * WW + sw) * CC + ch * 32;
        const unsigned doff = lds_offset(dl);
#pragma unroll
        for (int i = 0; i < 4; ++i) async_b128(doff + i * 16, src + i * 8);
      } else {
        u32x4 z = {0u, 0u, 0u, 0u};
        u32x4* d = (u32x4*)dl;
#pragma unroll
        for (int i = 0; i < 4; ++i) d[i] = z;
      }
    }
    wait_async0();
    __syncthreads();

    float acc[32] = {};
#pragma unroll
    for (int iy = 0; iy < 7; ++iy) {
#pragma unroll
      for (int ix = 0; ix < 7; ++ix) {
        const float p = s[iy * 7 + ix];
        if (p == 0.f) continue;  // masked / underflowed neighbors
        const h8* vp =
            (const h8*)(smem + ((ty + iy) * HALO_W + (tx + ix)) * VP);
#pragma unroll
        for (int i = 0; i < 4; ++i) {
          h8 t = vp[i];
#pragma unroll
          for (int e = 0; e < 8; ++e) acc[i * 8 + e] += p * (float)t[e];
        }
      }
    }
    GDEV _Float16* op = att_h + pix * CC + ch * 32;
#pragma unroll
    for (int v = 0; v < 4; ++v) {
      h8 o;
#pragma unroll
      for (int e = 0; e < 8; ++e) o[e] = (_Float16)(acc[v * 8 + e] * inv);
      ((GDEV h8*)op)[v] = o;
    }
  }
}

// ----------------------- Stage 3: output projection + ReLU ------------------
// One wave per 16-row M-tile: N=128 (8 tiles) x K=128 (4 chunks) -> 32 WMMAs.
__global__ __launch_bounds__(128) void proj_out_kernel(const _Float16* att_h_,
                                                       const _Float16* wvT_,
                                                       const float* bv_,
                                                       float* out_) {
  const GDEV _Float16* att_h = (const GDEV _Float16*)att_h_;
  const GDEV _Float16* wvT = (const GDEV _Float16*)wvT_;
  const GDEV float* bv = (const GDEV float*)bv_;
  GDEV float* out = (GDEV float*)out_;

  const int lane = threadIdx.x & 31;
  const int wave = blockIdx.x * 4 + (threadIdx.x >> 5);
  const int m0   = wave * 16;

  v8f acc[8] = {};
#pragma unroll
  for (int kc = 0; kc < 4; ++kc) {
    v16h a = load_a_f16(att_h + (size_t)m0 * CC + kc * 32, CC, lane);
#pragma unroll
    for (int nt = 0; nt < 8; ++nt) {
      v16h b = load_bT(wvT, CC, kc, nt, lane);
      acc[nt] = __builtin_amdgcn_wmma_f32_16x16x32_f16(
          false, a, false, b, (short)0, acc[nt], false, false);
    }
  }
  const int n = lane & 15, hi = lane >> 4;
#pragma unroll
  for (int nt = 0; nt < 8; ++nt) {
    float bias = bv[nt * 16 + n];
#pragma unroll
    for (int i = 0; i < 8; ++i) {
      float v = acc[nt][i] + bias;
      out[(size_t)(m0 + i + 8 * hi) * OD + nt * 16 + n] = v > 0.f ? v : 0.f;
    }
  }
}

// ----------------------- launch ---------------------------------------------
extern "C" void kernel_launch(void* const* d_in, const int* in_sizes, int n_in,
                              void* d_out, int out_size, void* d_ws,
                              size_t ws_size, hipStream_t stream) {
  const float* Q  = (const float*)d_in[0];
  const float* K  = (const float*)d_in[1];
  const float* V  = (const float*)d_in[2];
  const float* Wq = (const float*)d_in[3];
  const float* bq = (const float*)d_in[4];
  const float* Wk = (const float*)d_in[5];
  const float* bk = (const float*)d_in[6];
  const float* Wv = (const float*)d_in[7];
  const float* bv = (const float*)d_in[8];
  float* out = (float*)d_out;

  // Workspace layout (f16, 256B-aligned slices). Total ~14.3 MB.
  char* ws = (char*)d_ws;
  size_t off = 0;
  auto alloc = [&](size_t nhalf) {
    _Float16* p = (_Float16*)(ws + off);
    off += ((nhalf * sizeof(_Float16) + 255) / 256) * 256;
    return p;
  };
  _Float16* q_h   = alloc((size_t)MROWS * KD);
  _Float16* k_h   = alloc((size_t)MROWS * KD);
  _Float16* v_h   = alloc((size_t)MROWS * CC);
  _Float16* att_h = alloc((size_t)MROWS * CC);
  _Float16* wqT   = alloc((size_t)KD * CC);   // [64][128]
  _Float16* wkT   = alloc((size_t)KD * CC);   // [64][128]
  _Float16* wvT   = alloc((size_t)OD * CC);   // [128][128]
  (void)ws_size; (void)in_sizes; (void)n_in; (void)out_size;

  // Stage 0: weight transposition + f16 conversion, V -> f16.
  cvtT_f32_to_f16<<<(CC * KD + 255) / 256, 256, 0, stream>>>(Wq, wqT, CC, KD);
  cvtT_f32_to_f16<<<(CC * KD + 255) / 256, 256, 0, stream>>>(Wk, wkT, CC, KD);
  cvtT_f32_to_f16<<<(CC * OD + 255) / 256, 256, 0, stream>>>(Wv, wvT, CC, OD);
  cvt_f32_to_f16<<<((size_t)MROWS * CC + 255) / 256, 256, 0, stream>>>(
      V, v_h, MROWS * CC);

  // Stage 1: q/k projections (WMMA). 1152 M-tiles / 4 waves per block.
  proj_qk_kernel<<<MROWS / 16 / 4, 128, 0, stream>>>(Q, K, wqT, wkT, bq, bk,
                                                     q_h, k_h);

  // Stage 2: 7x7 neighborhood attention (async LDS staging).
  natten_kernel<<<dim3(WW / TW, HH / TH, BATCH), 128, 0, stream>>>(q_h, k_h,
                                                                   v_h, att_h);

  // Stage 3: output projection + bias + ReLU (WMMA).
  proj_out_kernel<<<MROWS / 16 / 4, 128, 0, stream>>>(att_h, wvT, bv, out);
}
